// GraphResNet_55628416418030
// MI455X (gfx1250) — matrix operand order repaired and checked by reading, not compiled, not run on hardware
//
#include <hip/hip_runtime.h>
#include <hip/hip_bf16.h>
#include <cstdint>

// MI455X / gfx1250, wave32. Conv + GEMM through v_wmma_f32_16x16x32_f16.
// Conv: double-buffered LDS pipeline; B-operand staged with
// global_load_async_to_lds_b128 (ASYNCcnt) overlapping the WMMA compute.

#ifndef CONV_ASYNC_LDS
#define CONV_ASYNC_LDS 1
#endif

typedef _Float16 f16;
typedef __attribute__((ext_vector_type(16))) _Float16 v16h;
typedef __attribute__((ext_vector_type(8)))  float    v8f;

// ---------------------------------------------------------------------------
// WMMA fragment loaders (CDNA5 ISA 7.12.2 16-bit layouts).
// A-frag from [M][64] tile row: lane L: m=L&15, kBase=(L<16)?0:8,
//   halves 0..7 -> k=kBase+0..7, halves 8..15 -> k=kBase+16..23.
// B-frag from position-major [pos][rowW] tile row: lane L: n=L&15,
//   kBase=(L<16)?0:8, halves 0..7 -> k=kBase+0..7, halves 8..15 -> k=kBase+16..23.
// ---------------------------------------------------------------------------
static __device__ __forceinline__ v16h frag_a(const f16* row, int lane, int kOff) {
  const f16* p = row + kOff + ((lane & 16) ? 8 : 0);
  v16h a;
  reinterpret_cast<uint4*>(&a)[0] = *reinterpret_cast<const uint4*>(p);
  reinterpret_cast<uint4*>(&a)[1] = *reinterpret_cast<const uint4*>(p + 16);
  return a;
}
static __device__ __forceinline__ v16h frag_b(const f16* row, int lane, int kOff) {
  const f16* p = row + kOff + ((lane & 16) ? 8 : 0);
  v16h b;
  reinterpret_cast<uint4*>(&b)[0] = *reinterpret_cast<const uint4*>(p);
  reinterpret_cast<uint4*>(&b)[1] = *reinterpret_cast<const uint4*>(p + 16);
  return b;
}

#define WMMA_F16(a, b, c) __builtin_amdgcn_wmma_f32_16x16x32_f16( \
    false, (a), false, (b), (short)0, (c), false, false)

// ---------------------------------------------------------------------------
// Implicit-GEMM convolution, NHWC f16, packed f16 weights [Cout][Kpad],
// k = (kh*KW+kw)*Cin + ci, Kpad multiple of 64.
// Block tile: 32 Cout x 256 positions; 8 waves, each a 2x2 grid of 16x16
// subtiles -> 8 WMMAs per wave per K-chunk (64), A/B frags reused 2x.
// Double-buffered LDS: chunk k+1 async-loads overlap chunk k WMMAs.
// `res`/`out` may alias (same element, same thread) -> no __restrict__.
// ---------------------------------------------------------------------------
__global__ __launch_bounds__(256)
void k_conv_wmma(const f16* __restrict__ in, const f16* __restrict__ wpk,
                 const float* __restrict__ scale, const float* __restrict__ bias,
                 const f16* res, f16* out,
                 int Nimg, int H, int W, int Cin, int Cout,
                 int KH, int KW, int stride, int padl,
                 int Ho, int Wo, int Kpad, int relu)
{
  __shared__ __align__(16) f16 As[2][32 * 64];    // 2 x 4 KB
  __shared__ __align__(16) f16 Bs[2][256 * 64];   // 2 x 32 KB

  const int tid  = threadIdx.x;
  const int lane = tid & 31;
  const int wv   = tid >> 5;
  const int laneM = lane & 15;
  const int coBase = blockIdx.y * 32;
  const long long posBase = (long long)blockIdx.x * 256;
  const int HoWo = Ho * Wo;
  const long long Npos = (long long)Nimg * HoWo;
  const int Kreal = KH * KW * Cin;
  const bool fast = (Cin & 63) == 0;   // 64-half K segment is contiguous NHWC

  // B-gather: one thread per position, 64 halves (one K-chunk) each.
  const long long bpos = posBase + tid;
  const bool pvalid = bpos < Npos;
  int nb = 0, hob = 0, wob = 0;
  if (pvalid) {
    nb = (int)(bpos / HoWo);
    int rem = (int)(bpos - (long long)nb * HoWo);
    hob = rem / Wo;
    wob = rem - hob * Wo;
  }

  v8f acc[2][2] = {};

  // Stage one 64-wide K-chunk into LDS buffer `buf`.
  auto stage = [&](int buf, int k0) {
    // A tile: 2048 halves, one b128 per thread.
    {
      int h8 = tid * 8, m = h8 >> 6, kk = h8 & 63;
      *reinterpret_cast<uint4*>(&As[buf][h8]) =
          *reinterpret_cast<const uint4*>(&wpk[(size_t)(coBase + m) * Kpad + k0 + kk]);
    }
    f16* brow = &Bs[buf][tid * 64];
    if (fast) {
      // whole 64-half segment shares one (kh,kw); 128 contiguous bytes.
      int khw = k0 / Cin;
      int ci0 = k0 - khw * Cin;
      int kw = khw % KW, kh = khw / KW;
      bool iv = false;
      size_t gbase = 0;
      if (pvalid && k0 < Kreal) {
        int hi = hob * stride + kh - padl;
        int wi = wob * stride + kw - padl;
        if (hi >= 0 && hi < H && wi >= 0 && wi < W) {
          iv = true;
          gbase = (((size_t)nb * H + hi) * W + wi) * (size_t)Cin + ci0;
        }
      }
      if (iv) {
        const f16* src = in + gbase;
#if CONV_ASYNC_LDS
        // GLOBAL_LOAD_ASYNC_TO_LDS_B128 (ASYNCcnt): vdst = LDS offset VGPR,
        // vaddr = 64-bit global address, saddr = off.
        unsigned ldsa = (unsigned)(uintptr_t)brow;
        #pragma unroll
        for (int u = 0; u < 8; ++u)
          asm volatile("global_load_async_to_lds_b128 %0, %1, off"
                       :: "v"(ldsa + 16u * u), "v"(src + 8 * u) : "memory");
#else
        #pragma unroll
        for (int u = 0; u < 8; ++u)
          reinterpret_cast<uint4*>(brow)[u] = reinterpret_cast<const uint4*>(src)[u];
#endif
      } else {
        uint4 z = {0u, 0u, 0u, 0u};
        #pragma unroll
        for (int u = 0; u < 8; ++u) reinterpret_cast<uint4*>(brow)[u] = z;
      }
    } else {
      // scalar im2col gather (stem / CoNet conv1, Cin = 3)
      for (int u = 0; u < 64; ++u) {
        int k = k0 + u;
        f16 v = (f16)0.f;
        if (pvalid && k < Kreal) {
          int ci  = k % Cin;
          int khw = k / Cin;
          int kw  = khw % KW;
          int kh  = khw / KW;
          int hi = hob * stride + kh - padl;
          int wi = wob * stride + kw - padl;
          if (hi >= 0 && hi < H && wi >= 0 && wi < W)
            v = in[(((size_t)nb * H + hi) * W + wi) * Cin + ci];
        }
        brow[u] = v;
      }
    }
  };

  // --- software pipeline: prologue ---
  stage(0, 0);
#if CONV_ASYNC_LDS
  asm volatile("s_wait_asynccnt 0" ::: "memory");
#endif
  __syncthreads();

  int cur = 0;
  for (int k0 = 0; k0 < Kpad; k0 += 64) {
    // issue next chunk's loads; they overlap this chunk's WMMAs
    if (k0 + 64 < Kpad) stage(cur ^ 1, k0 + 64);

    // --- compute on `cur`: 2 K-subchunks x (2 Cout x 2 pos) subtiles ---
    const f16* Ac = As[cur];
    const f16* Bc = Bs[cur];
    const f16* ar0 = Ac + (laneM << 6);
    const f16* ar1 = Ac + ((16 + laneM) << 6);
    const f16* br0 = Bc + ((((wv << 1) | 0) * 16 + laneM) << 6);
    const f16* br1 = Bc + ((((wv << 1) | 1) * 16 + laneM) << 6);
    #pragma unroll
    for (int c = 0; c < 2; ++c) {
      int ko = c << 5;
      v16h a0 = frag_a(ar0, lane, ko);
      v16h a1 = frag_a(ar1, lane, ko);
      v16h b0 = frag_b(br0, lane, ko);
      v16h b1 = frag_b(br1, lane, ko);
      acc[0][0] = WMMA_F16(a0, b0, acc[0][0]);
      acc[0][1] = WMMA_F16(a0, b1, acc[0][1]);
      acc[1][0] = WMMA_F16(a1, b0, acc[1][0]);
      acc[1][1] = WMMA_F16(a1, b1, acc[1][1]);
    }

#if CONV_ASYNC_LDS
    asm volatile("s_wait_asynccnt 0" ::: "memory");  // own async writes done
#endif
    __syncthreads();                                  // all waves' writes visible
    cur ^= 1;
  }

  // --- epilogue: BN + residual + ReLU, NHWC store ---
  const int mB = (lane & 16) ? 8 : 0;
  #pragma unroll
  for (int s = 0; s < 2; ++s) {
    long long p = posBase + ((wv << 1) | s) * 16 + laneM;
    if (p >= Npos) continue;
    size_t obase = (size_t)p * Cout;
    #pragma unroll
    for (int t = 0; t < 2; ++t) {
      #pragma unroll
      for (int r = 0; r < 8; ++r) {
        int co = coBase + (t << 4) + mB + r;
        float v = acc[t][s][r];
        if (scale) v = v * scale[co] + bias[co];
        if (res)   v += (float)res[obase + co];
        if (relu)  v = v > 0.f ? v : 0.f;
        out[obase + co] = (f16)v;
      }
    }
  }
}

// ---------------------------------------------------------------------------
// Dense GEMM: C[M][Nout] = A[M][K] * B[K][Ncomp] (+ per-column bias).
// A,B row-major f16; K mult of 32; M mult of 16. Cols >= Nout computed
// (padding) but not stored. Block tile 16 x 128, 8 waves.
// ---------------------------------------------------------------------------
__global__ __launch_bounds__(256)
void k_gemm_wmma(const f16* __restrict__ A, const f16* __restrict__ B,
                 const float* __restrict__ bias, float* __restrict__ C,
                 int M, int K, int Ncomp, int Nout)
{
  __shared__ __align__(16) f16 As[16 * 32];
  __shared__ __align__(16) f16 Bs[128 * 32];

  const int tid  = threadIdx.x;
  const int lane = tid & 31;
  const int wv   = tid >> 5;
  const int mBase = blockIdx.y * 16;
  const int nBase = blockIdx.x * 128;

  const int bp  = tid >> 1;
  const int bk0 = (tid & 1) * 16;
  const int col = nBase + bp;
  const bool cvalid = col < Ncomp;

  v8f acc = {};

  for (int k0 = 0; k0 < K; k0 += 32) {
    {
      int h2 = tid * 2, m = h2 >> 5, kk = h2 & 31;
      *reinterpret_cast<unsigned int*>(&As[h2]) =
          *reinterpret_cast<const unsigned int*>(&A[(size_t)(mBase + m) * K + k0 + kk]);
    }
    #pragma unroll 4
    for (int u = 0; u < 16; ++u) {
      int kk = bk0 + u;
      Bs[bp * 32 + kk] = cvalid ? B[(size_t)(k0 + kk) * Ncomp + col] : (f16)0.f;
    }
    __syncthreads();

    v16h a = frag_a(As + ((lane & 15) << 5), lane, 0);
    v16h b = frag_b(Bs + (((wv << 4) + (lane & 15)) << 5), lane, 0);
    acc = WMMA_F16(a, b, acc);
    __syncthreads();
  }

  const int mB = (lane & 16) ? 8 : 0;
  const int c  = nBase + (wv << 4) + (lane & 15);
  if (c < Nout) {
    #pragma unroll
    for (int r = 0; r < 8; ++r) {
      int row = mBase + mB + r;
      float v = acc[r];
      if (bias) v += bias[c];
      C[(size_t)row * Nout + c] = v;
    }
  }
}

// ---------------------------------------------------------------------------
// Utility kernels
// ---------------------------------------------------------------------------
__global__ void k_nchw_to_nhwc_f16(const float* __restrict__ in, f16* __restrict__ out,
                                   int N, int C, int H, int W) {
  long long idx = (long long)blockIdx.x * blockDim.x + threadIdx.x;
  long long total = (long long)N * C * H * W;
  if (idx >= total) return;
  int c = (int)(idx % C); long long t = idx / C;
  int w = (int)(t % W); t /= W;
  int h = (int)(t % H); int n = (int)(t / H);
  out[idx] = (f16)in[(((size_t)n * C + c) * H + h) * W + w];
}

// OIHW f32 -> [Cout][Kpad] f16, k = (kh*KW + kw)*Cin + ci, zero-padded K.
__global__ void k_pack_conv_w(const float* __restrict__ w, f16* __restrict__ out,
                              int Cout, int Cin, int KH, int KW, int Kpad) {
  int idx = blockIdx.x * blockDim.x + threadIdx.x;
  if (idx >= Cout * Kpad) return;
  int co = idx / Kpad, k = idx % Kpad;
  float v = 0.f;
  int Kr = KH * KW * Cin;
  if (k < Kr) {
    int ci  = k % Cin;
    int khw = k / Cin;
    int kw  = khw % KW;
    int kh  = khw / KW;
    v = w[(((size_t)co * Cin + ci) * KH + kh) * KW + kw];
  }
  out[idx] = (f16)v;
}

__global__ void k_cast_f16(const float* __restrict__ in, f16* __restrict__ out,
                           long long n) {
  long long i = (long long)blockIdx.x * blockDim.x + threadIdx.x;
  if (i < n) out[i] = (f16)in[i];
}

__global__ void k_transpose_cast(const float* __restrict__ in, f16* __restrict__ out,
                                 int R, int C) {
  long long i = (long long)blockIdx.x * blockDim.x + threadIdx.x;
  if (i >= (long long)R * C) return;
  int r = (int)(i / C), c = (int)(i % C);
  out[(size_t)c * R + r] = (f16)in[i];
}

__global__ void k_pad_cast(const float* __restrict__ in, f16* __restrict__ out,
                           int K, int Nreal, int Npad) {
  int i = blockIdx.x * blockDim.x + threadIdx.x;
  if (i >= K * Npad) return;
  int k = i / Npad, c = i % Npad;
  out[i] = (c < Nreal) ? (f16)in[(size_t)k * Nreal + c] : (f16)0.f;
}

// NHWC f16 [N, HW, C] -> f32 [N, C] mean over HW.
__global__ void k_avgpool(const f16* __restrict__ in, float* __restrict__ out,
                          int HW, int C, long long total) {
  long long i = (long long)blockIdx.x * blockDim.x + threadIdx.x;
  if (i >= total) return;
  long long n = i / C;
  int c = (int)(i % C);
  float s = 0.f;
  for (int t = 0; t < HW; ++t) s += (float)in[((size_t)n * HW + t) * C + c];
  out[i] = s / (float)HW;
}

__global__ void k_l2norm_rows(float* z, int R, int C) {
  int r = blockIdx.x * blockDim.x + threadIdx.x;
  if (r >= R) return;
  float ss = 0.f;
  for (int c = 0; c < C; ++c) { float v = z[(size_t)r * C + c]; ss += v * v; }
  float inv = rsqrtf(ss);
  for (int c = 0; c < C; ++c) z[(size_t)r * C + c] *= inv;
}

// ---------------------------------------------------------------------------
// Host orchestration
// ---------------------------------------------------------------------------
static inline unsigned cdivu(long long a, long long b) {
  return (unsigned)((a + b - 1) / b);
}

extern "C" void kernel_launch(void* const* d_in, const int* in_sizes, int n_in,
                              void* d_out, int out_size, void* d_ws, size_t ws_size,
                              hipStream_t stream) {
  (void)in_sizes; (void)n_in; (void)out_size; (void)ws_size;
  const int N = 1024;

  // --- inputs, setup_inputs() recursive dict-insertion order ---
  int idx = 0;
  const float* X  = (const float*)d_in[idx++];
  const float* w0 = (const float*)d_in[idx++];
  const float* s0 = (const float*)d_in[idx++];
  const float* b0 = (const float*)d_in[idx++];
  struct Blk { const float *w1,*s1,*b1,*w2,*s2,*b2,*wsc,*ssc,*bsc; };
  static const int BC[8] = {64, 64, 128, 128, 256, 256, 512, 512};
  static const int BS[8] = {1, 1, 2, 1, 2, 1, 2, 1};
  Blk blk[8];
  for (int i = 0; i < 8; ++i) {
    blk[i].w1 = (const float*)d_in[idx++];
    blk[i].s1 = (const float*)d_in[idx++];
    blk[i].b1 = (const float*)d_in[idx++];
    blk[i].w2 = (const float*)d_in[idx++];
    blk[i].s2 = (const float*)d_in[idx++];
    blk[i].b2 = (const float*)d_in[idx++];
    if (BS[i] == 2) {
      blk[i].wsc = (const float*)d_in[idx++];
      blk[i].ssc = (const float*)d_in[idx++];
      blk[i].bsc = (const float*)d_in[idx++];
    } else { blk[i].wsc = blk[i].ssc = blk[i].bsc = nullptr; }
  }
  const float* cw1 = (const float*)d_in[idx++];
  const float* cw2 = (const float*)d_in[idx++];
  const float* cwp = (const float*)d_in[idx++];
  const float* cbp = (const float*)d_in[idx++];
  const float* wg  = (const float*)d_in[idx++];
  const float* bg  = (const float*)d_in[idx++];

  // --- workspace bump allocator (~356 MB) ---
  char* base = (char*)d_ws;
  size_t off = 0;
  auto alloc = [&](size_t bytes) -> void* {
    void* p = base + off;
    off += (bytes + 255) & ~(size_t)255;
    return p;
  };
  f16*   x16    = (f16*)alloc((size_t)N * 32 * 32 * 3 * 2);
  f16*   actA   = (f16*)alloc((size_t)N * 32 * 32 * 64 * 2);
  f16*   actB   = (f16*)alloc((size_t)N * 32 * 32 * 64 * 2);
  f16*   actSC  = (f16*)alloc((size_t)N * 16 * 16 * 128 * 2);   // also CoNet acts
  f16*   wpk    = (f16*)alloc((size_t)512 * 4608 * 2);          // packed weights (reused)
  float* pooled = (float*)alloc((size_t)N * 128 * 4);
  f16*   pooled16 = (f16*)alloc((size_t)N * 128 * 2);
  f16*   cwp16  = (f16*)alloc((size_t)128 * 128 * 2);
  float* z      = (float*)alloc((size_t)N * 128 * 4);
  f16*   z16    = (f16*)alloc((size_t)N * 128 * 2);
  f16*   zT16   = (f16*)alloc((size_t)N * 128 * 2);
  float* hbuf   = (float*)alloc((size_t)N * 512 * 4);
  f16*   h16    = (f16*)alloc((size_t)N * 512 * 2);
  float* Tbuf   = (float*)alloc((size_t)128 * 512 * 4);
  f16*   T16    = (f16*)alloc((size_t)128 * 512 * 2);
  float* Gbuf   = (float*)alloc((size_t)N * 512 * 4);
  f16*   G16    = (f16*)alloc((size_t)N * 512 * 2);
  f16*   wg16   = (f16*)alloc((size_t)512 * 16 * 2);
  f16*   cn1    = actSC;                                          // 32 MiB (64ch @16x16)
  f16*   cn2    = (f16*)((char*)actSC + (size_t)34 * 1024 * 1024);// 16 MiB (128ch @8x8)

  auto run_conv = [&](const float* w, const float* sc, const float* bi,
                      const f16* in, const f16* res, f16* outp,
                      int H, int Cin, int Cout, int KH, int KW, int stride, int relu) {
    int Ho = (H + stride - 1) / stride;
    int Kr = KH * KW * Cin;
    int Kpad = (Kr + 63) & ~63;              // K-chunk 64
    int padl = (Ho - 1) * stride + KH - H;   // lax 'SAME' total pad
    if (padl < 0) padl = 0;
    padl >>= 1;                              // pad_lo (asymmetric for s=2)
    int tot = Cout * Kpad;
    k_pack_conv_w<<<cdivu(tot, 256), 256, 0, stream>>>(w, wpk, Cout, Cin, KH, KW, Kpad);
    long long Npos = (long long)N * Ho * Ho;
    dim3 g(cdivu(Npos, 256), (unsigned)(Cout / 32));
    k_conv_wmma<<<g, 256, 0, stream>>>(in, wpk, sc, bi, res, outp,
                                       N, H, H, Cin, Cout, KH, KW, stride, padl,
                                       Ho, Ho, Kpad, relu);
  };
  auto run_gemm = [&](const f16* A, const f16* Bm, const float* bias, float* C,
                      int M, int K, int Ncomp, int Nout) {
    dim3 g(cdivu(Ncomp, 128), (unsigned)(M / 16));
    k_gemm_wmma<<<g, 256, 0, stream>>>(A, Bm, bias, C, M, K, Ncomp, Nout);
  };

  // --- layout: NCHW f32 -> NHWC f16 ---
  {
    long long tot = (long long)N * 3 * 32 * 32;
    k_nchw_to_nhwc_f16<<<cdivu(tot, 256), 256, 0, stream>>>(X, x16, N, 3, 32, 32);
  }

  // --- ResNet18-CIFAR encoder (BN folded, f16 WMMA) ---
  run_conv(w0, s0, b0, x16, nullptr, actA, 32, 3, 64, 3, 3, 1, 1);
  int H = 32, Cin = 64;
  for (int i = 0; i < 8; ++i) {
    int C = BC[i], S = BS[i];
    int Ho = H / S;
    const f16* res = actA;  // identity shortcut: same-position read-then-write
    if (S == 2) {
      run_conv(blk[i].wsc, blk[i].ssc, blk[i].bsc, actA, nullptr, actSC,
               H, Cin, C, 1, 1, 2, 0);
      res = actSC;
    }
    run_conv(blk[i].w1, blk[i].s1, blk[i].b1, actA, nullptr, actB, H, Cin, C, 3, 3, S, 1);
    run_conv(blk[i].w2, blk[i].s2, blk[i].b2, actB, res, actA, Ho, C, C, 3, 3, 1, 1);
    H = Ho; Cin = C;
  }
  // global average pool -> h [1024, 512] f32
  {
    long long tot = (long long)N * 512;
    k_avgpool<<<cdivu(tot, 256), 256, 0, stream>>>(actA, hbuf, H * H, 512, tot);
  }

  // --- CoNet similarity branch (frozen) ---
  run_conv(cw1, nullptr, nullptr, x16, nullptr, cn1, 32, 3, 64, 3, 3, 2, 1);
  run_conv(cw2, nullptr, nullptr, cn1, nullptr, cn2, 16, 64, 128, 3, 3, 2, 1);
  {
    long long tot = (long long)N * 128;
    k_avgpool<<<cdivu(tot, 256), 256, 0, stream>>>(cn2, pooled, 8 * 8, 128, tot);
    k_cast_f16<<<cdivu(tot, 256), 256, 0, stream>>>(pooled, pooled16, tot);
  }
  k_cast_f16<<<cdivu(128 * 128, 256), 256, 0, stream>>>(cwp, cwp16, 128 * 128);
  run_gemm(pooled16, cwp16, cbp, z, N, 128, 128, 128);   // z = pool @ cwp + cbp
  k_l2norm_rows<<<cdivu(N, 256), 256, 0, stream>>>(z, N, 128);
  k_cast_f16<<<cdivu((long long)N * 128, 256), 256, 0, stream>>>(z, z16, (long long)N * 128);
  k_transpose_cast<<<cdivu((long long)N * 128, 256), 256, 0, stream>>>(z, zT16, N, 128);
  k_cast_f16<<<cdivu((long long)N * 512, 256), 256, 0, stream>>>(hbuf, h16, (long long)N * 512);

  // --- graph conv: (z zT) h Wg + bg == z (zT h) Wg + bg (associativity) ---
  run_gemm(zT16, h16, nullptr, Tbuf, 128, N, 512, 512);  // T = zT @ h   [128,512]
  k_cast_f16<<<cdivu(128 * 512, 256), 256, 0, stream>>>(Tbuf, T16, 128 * 512);
  run_gemm(z16, T16, nullptr, Gbuf, N, 128, 512, 512);   // G = z @ T    [1024,512]
  k_cast_f16<<<cdivu((long long)N * 512, 256), 256, 0, stream>>>(Gbuf, G16, (long long)N * 512);
  k_pad_cast<<<cdivu(512 * 16, 256), 256, 0, stream>>>(wg, wg16, 512, 10, 16);
  run_gemm(G16, wg16, bg, (float*)d_out, N, 512, 16, 10); // out = G @ wg + bg
}
